// DNLCLayer_5488968205021
// MI455X (gfx1250) — matrix operand order repaired and checked by reading, not compiled, not run on hardware
//
#include <hip/hip_runtime.h>
#include <hip/hip_fp16.h>

typedef __attribute__((ext_vector_type(16))) _Float16 v16h;
typedef __attribute__((ext_vector_type(8)))  _Float16 v8h;
typedef __attribute__((ext_vector_type(8)))  float    v8f;
typedef __attribute__((ext_vector_type(4)))  int      v4i;

#define AS1 __attribute__((address_space(1)))
#define AS3 __attribute__((address_space(3)))

#define Bn   32
#define Cn   512
#define Hn   112
#define Wn   112
#define RCn  32
#define CPGn 16
#define OH   56
#define OW   56
#define Sdim 3136   // 56*56

#ifndef __has_builtin
#define __has_builtin(x) 0
#endif
#if __has_builtin(__builtin_amdgcn_global_load_async_to_lds_b128)
#define USE_ASYNC_LDS 1
#else
#define USE_ASYNC_LDS 0
#endif

__device__ __forceinline__ void wait_async_lds() {
#if USE_ASYNC_LDS
#if __has_builtin(__builtin_amdgcn_s_wait_asynccnt)
    __builtin_amdgcn_s_wait_asynccnt(0);
#else
    asm volatile("s_wait_asynccnt 0" ::: "memory");
#endif
#endif
}

// ---------------------------------------------------------------------------
// Kernel 0: zero the GAP accumulator (graph-capture safe init each launch)
// ---------------------------------------------------------------------------
__global__ void zero_f32(float* __restrict__ p, int n) {
    int i = blockIdx.x * blockDim.x + threadIdx.x;
    if (i < n) p[i] = 0.0f;
}

// ---------------------------------------------------------------------------
// Kernel 1: fused grouped 3x3/s2 conv (theta & phi share x reads) + GAP·fc_w.
// One block = (batch b, group rc, 4 output rows). 256 threads (8 waves).
// Input tile (9 rows x 112 cols) staged to LDS in two 8-channel phases via
// coalesced b128 async-to-LDS DMA (ASYNCcnt path). Conv + GAP read from LDS;
// each input byte crosses HBM exactly once.
// ---------------------------------------------------------------------------
__global__ __launch_bounds__(256)
void fused_conv_gap(const float* __restrict__ x,
                    const float* __restrict__ theta_w, const float* __restrict__ theta_b,
                    const float* __restrict__ phi_w,   const float* __restrict__ phi_b,
                    const float* __restrict__ fc_w,
                    _Float16* __restrict__ theta_h, _Float16* __restrict__ phi_h,
                    float* __restrict__ gsum)
{
    __shared__ float tile[8 * 9 * Wn];   // 8 channels x 9 input rows x 112 cols
    __shared__ float wt[CPGn * 9];
    __shared__ float wp[CPGn * 9];
    __shared__ float fcw[CPGn];
    __shared__ float red[256];

    const int tid = threadIdx.x;
    const int rt  = blockIdx.x;   // 0..13 : tile of 4 output rows
    const int rc  = blockIdx.y;   // group / reduction channel
    const int b   = blockIdx.z;

    if (tid < CPGn * 9) {
        wt[tid] = theta_w[rc * CPGn * 9 + tid];
        wp[tid] = phi_w  [rc * CPGn * 9 + tid];
    }
    if (tid < CPGn) fcw[tid] = fc_w[rc * CPGn + tid];

    const int r0 = rt * 4;
    const int r  = r0 + tid / OW;   // valid for tid < 224
    const int c  = tid % OW;

    const float* __restrict__ xg =
        x + (size_t)(b * Cn + rc * CPGn) * (Hn * Wn);

    float at = 0.0f, ap = 0.0f, ga = 0.0f;

    for (int ph = 0; ph < 2; ++ph) {
        const int ch0 = ph * 8;

        // ---- stage 8 channels x 9 rows x 112 cols into LDS (b128) ----
        // 8*9*28 = 2016 float4 elements
        for (int j = tid; j < 2016; j += 256) {
            const int chl = j / 252;            // 9*28
            const int rem = j - chl * 252;
            const int ry  = rem / 28;
            const int rx  = rem - ry * 28;
            const int iy  = 2 * r0 - 1 + ry;
            float* ldst = &tile[(chl * 9 + ry) * Wn + rx * 4];
            if ((unsigned)iy < (unsigned)Hn) {
                const float* gsrc = xg + (size_t)(ch0 + chl) * (Hn * Wn)
                                       + iy * Wn + rx * 4;
#if USE_ASYNC_LDS
                __builtin_amdgcn_global_load_async_to_lds_b128(
                    (AS1 v4i*)gsrc, (AS3 v4i*)ldst, 0, 0);
#else
                *(float4*)ldst = *(const float4*)gsrc;
#endif
            } else {
                *(float4*)ldst = make_float4(0.f, 0.f, 0.f, 0.f);
            }
        }
        wait_async_lds();
        __syncthreads();

        // ---- conv accumulate from LDS (tid < 224: one output pixel) ----
        if (tid < 224) {
            const int ryb = 2 * (r - r0);       // 0,2,4,6
            #pragma unroll
            for (int chl = 0; chl < 8; ++chl) {
                const float* __restrict__ w0 = &wt[(ch0 + chl) * 9];
                const float* __restrict__ w1 = &wp[(ch0 + chl) * 9];
                const float* __restrict__ trow = &tile[chl * 9 * Wn];
                #pragma unroll
                for (int ky = 0; ky < 3; ++ky) {
                    const float* __restrict__ tr = trow + (ryb + ky) * Wn;
                    #pragma unroll
                    for (int kx = 0; kx < 3; ++kx) {
                        const int ix = 2 * c - 1 + kx;   // [-1, 111]
                        if (ix >= 0) {
                            const float v = tr[ix];
                            at = fmaf(v, w0[ky * 3 + kx], at);
                            ap = fmaf(v, w1[ky * 3 + kx], ap);
                        }
                    }
                }
            }
        }

        // ---- GAP partial from LDS: owned input rows = tile rows 1..8 ----
        // 8 ch x 8 rows x 112 cols = 7168 elements
        for (int i = tid; i < 7168; i += 256) {
            const int chl = i / 896;
            const int rem = i - chl * 896;
            const int ry  = 1 + rem / Wn;
            const int ix  = rem - (ry - 1) * Wn;
            ga += tile[(chl * 9 + ry) * Wn + ix] * fcw[ch0 + chl];
        }
        __syncthreads();   // tile reused next phase
    }

    if (tid < 224) {
        const size_t so = (size_t)(b * RCn + rc) * Sdim + r * OW + c;
        theta_h[so] = (_Float16)(at + theta_b[rc]);
        phi_h[so]   = (_Float16)(ap + phi_b[rc]);
    }

    // ---- block reduction of GAP partial, one atomic per block ----
    red[tid] = ga;
    __syncthreads();
    if (tid < 32) {
        float s = 0.0f;
        #pragma unroll
        for (int k = 0; k < 8; ++k) s += red[tid + 32 * k];
        red[tid] = s;
    }
    __syncthreads();
    if (tid == 0) {
        float s = 0.0f;
        #pragma unroll
        for (int i = 0; i < 32; ++i) s += red[i];
        atomicAdd(&gsum[b * RCn + rc], s);
    }
}

// ---------------------------------------------------------------------------
// Kernel 2: per-batch attention. mat = theta (32xS) * phi^T (Sx32) via
// v_wmma_f32_16x16x32_f16 (4 waves, one 16x16 tile each, 98 K-steps),
// then row softmax + dot with g in LDS.  (codegen verified round 1)
// ---------------------------------------------------------------------------
__global__ __launch_bounds__(128)
void attn_wmma(const _Float16* __restrict__ theta_h,
               const _Float16* __restrict__ phi_h,
               const float* __restrict__ gsum,
               const float* __restrict__ fc_b,
               float* __restrict__ out)
{
    __shared__ float mat[RCn * RCn];
    __shared__ float gv[RCn];

    const int b    = blockIdx.x;
    const int tid  = threadIdx.x;
    const int wave = tid >> 5;
    const int lane = tid & 31;
    const int mi   = wave >> 1;   // tile row (0/1)
    const int ni   = wave & 1;    // tile col (0/1)

    // A fragment (16-bit A 16x32, ISA 7.12.2): lanes<16 hold K {0..7,16..23},
    // lanes>=16 hold K {8..15,24..31}; M = lane&15 within the tile.
    const _Float16* __restrict__ Aptr =
        theta_h + (size_t)(b * RCn + mi * 16 + (lane & 15)) * Sdim
                + ((lane >= 16) ? 8 : 0);
    // B fragment (32x16 = phi^T): N = lane&15; lanes<16 -> K {0..7,16..23},
    // lanes>=16 -> K {8..15,24..31} (same K order as A).
    const _Float16* __restrict__ Bptr =
        phi_h + (size_t)(b * RCn + ni * 16 + (lane & 15)) * Sdim
              + ((lane >= 16) ? 8 : 0);

    v8f acc = {};
    for (int k0 = 0; k0 < Sdim; k0 += 32) {
        union { v16h v; v8h h[2]; } a, bm;
        a.h[0]  = *(const v8h*)(Aptr + k0);        // K chunk base..+7
        a.h[1]  = *(const v8h*)(Aptr + k0 + 16);   // K chunk base+16..+23
        bm.h[0] = *(const v8h*)(Bptr + k0);        // K base..+7
        bm.h[1] = *(const v8h*)(Bptr + k0 + 16);   // K base+16..+23
        acc = __builtin_amdgcn_wmma_f32_16x16x32_f16(
                  /*neg_a=*/false, a.v, /*neg_b=*/false, bm.v,
                  /*c_mod=*/(short)0, acc, /*reuse_a=*/false, /*reuse_b=*/false);
    }

    // Scatter C tile to LDS (32-bit C 16x16 layout: VGPR j -> M=j / M=j+8)
    #pragma unroll
    for (int j = 0; j < 8; ++j) {
        const int row = mi * 16 + ((lane < 16) ? j : j + 8);
        const int col = ni * 16 + (lane & 15);
        mat[row * RCn + col] = acc[j];
    }
    if (tid < RCn)
        gv[tid] = gsum[b * RCn + tid] * (1.0f / (float)(Hn * Wn)) + fc_b[tid];
    __syncthreads();

    if (tid < RCn) {
        const int row = tid;
        float m = -3.402823466e38f;
        #pragma unroll
        for (int d = 0; d < RCn; ++d) m = fmaxf(m, mat[row * RCn + d]);
        float e[RCn];
        float s = 0.0f;
        #pragma unroll
        for (int d = 0; d < RCn; ++d) { e[d] = __expf(mat[row * RCn + d] - m); s += e[d]; }
        const float inv = 1.0f / s;
        float o = 0.0f;
        #pragma unroll
        for (int d = 0; d < RCn; ++d) o += e[d] * inv * gv[d];
        out[b * RCn + row] = o;
    }
}

// ---------------------------------------------------------------------------
extern "C" void kernel_launch(void* const* d_in, const int* in_sizes, int n_in,
                              void* d_out, int out_size, void* d_ws, size_t ws_size,
                              hipStream_t stream)
{
    const float* x       = (const float*)d_in[0];
    const float* theta_w = (const float*)d_in[1];
    const float* theta_b = (const float*)d_in[2];
    const float* phi_w   = (const float*)d_in[3];
    const float* phi_b   = (const float*)d_in[4];
    const float* fc_w    = (const float*)d_in[5];
    const float* fc_b    = (const float*)d_in[6];
    float* out = (float*)d_out;

    // Workspace layout: [gsum: B*RC f32 | theta_h: B*RC*S f16 | phi_h: ...]
    float*    gsum    = (float*)d_ws;
    _Float16* theta_h = (_Float16*)((char*)d_ws + 4096);
    _Float16* phi_h   = theta_h + (size_t)Bn * RCn * Sdim;

    zero_f32<<<(Bn * RCn + 255) / 256, 256, 0, stream>>>(gsum, Bn * RCn);

    fused_conv_gap<<<dim3(OH / 4, RCn, Bn), 256, 0, stream>>>(
        x, theta_w, theta_b, phi_w, phi_b, fc_w, theta_h, phi_h, gsum);

    attn_wmma<<<Bn, 128, 0, stream>>>(theta_h, phi_h, gsum, fc_b, out);
}
